// Brain_7035156431411
// MI455X (gfx1250) — matrix operand order repaired and checked by reading, not compiled, not run on hardware
//
#include <hip/hip_runtime.h>
#include <math.h>

// ---- problem constants (from reference) ----
#define T_  8
#define B_  32
#define D_  768
#define M_  8
#define S_  2
#define L_  2
#define H_  8
#define d_  384         // D/S
#define P_  16          // M*S
#define HD_ 96          // D/H

// ---- WMMA types ----
typedef __attribute__((ext_vector_type(16))) __bf16 bf16x16;
typedef __attribute__((ext_vector_type(2)))  __bf16 bf16x2;
typedef __attribute__((ext_vector_type(8)))  float  f32x8;

__device__ __forceinline__ unsigned short f2bf(float f) {
  unsigned int u = __float_as_uint(f);
  u += 0x7FFFu + ((u >> 16) & 1u);      // round-to-nearest-even
  return (unsigned short)(u >> 16);
}
__device__ __forceinline__ unsigned int pack2bf(float lo, float hi) {
#if __has_builtin(__builtin_amdgcn_cvt_pk_bf16_f32)
  union { bf16x2 v; unsigned int u; } cv;
  cv.v = __builtin_amdgcn_cvt_pk_bf16_f32(lo, hi);   // v_cvt_pk_bf16_f32
  return cv.u;
#else
  return (unsigned int)f2bf(lo) | ((unsigned int)f2bf(hi) << 16);
#endif
}

// =====================================================================
// Generic batched GEMM (exact tiling REQUIRED: Mr%32==0, N%128==0, K%32==0):
//   C[z][m][n] = ( sum_k A[z][m][k] * B[z][k][n]  + bias[z][n] ) * alpha
//   A element: AT ? A[k*lda+m] : A[m*lda+k]
//   B element: WT ? W[n*ldw+k] : W[k*ldw+n]
// 256 threads = 8 wave32 waves; block tile 32(M) x 128(N), K-step 32.
// Each wave owns two 16x16 C tiles (n and n+64): 2 v_wmma per K-chunk.
// LDS double-buffered; bf16 pairs packed into b32; contiguous pairs use
// float2 (global_load_b64). Software pipeline: loads for chunk k+1 issued
// before the WMMAs of chunk k; one barrier per chunk.
// =====================================================================
#define TBM 32
#define TBN 128
#define TBK 32
#define SA_STRIDE 18    // uints per A row   (16 + 2 pad)
#define SB_STRIDE 130   // uints per B k2row (128 + 2 pad)

template<int AT, int WT>
__global__ __launch_bounds__(256) void wmma_gemm(
    const float* __restrict__ A, long sAz, int lda,
    const float* __restrict__ W, long sWz, int ldw,
    const float* __restrict__ bias, long sBz,
    float* __restrict__ C, long sCz, int ldc,
    int K, float alpha, int relu)
{
  __shared__ unsigned int sA[2][TBM * SA_STRIDE];
  __shared__ unsigned int sB[2][(TBK / 2) * SB_STRIDE];

  const float* Az = A + (size_t)blockIdx.z * (size_t)sAz;
  const float* Wz = W + (size_t)blockIdx.z * (size_t)sWz;
  float*       Cz = C + (size_t)blockIdx.z * (size_t)sCz;

  const int rowBase = blockIdx.y * TBM;
  const int colBase = blockIdx.x * TBN;
  const int tid  = threadIdx.x;
  const int lane = tid & 31;
  const int wv   = tid >> 5;
  const int wRow = (wv >> 2) * 16;   // 0 or 16
  const int wCol = (wv & 3) * 16;    // 0,16,32,48  (+64 for second tile)

  // staging decomposition (A: 32x16 uints, 2/thread; B: 16x128 uints, 8/thread)
  const int ar  = tid >> 4;          // 0..15 (+16 on 2nd elem)
  const int ac2 = tid & 15;          // packed-k column
  const int bk2 = tid >> 7;          // 0..1  (+2 per elem)
  const int bn  = tid & 127;

  f32x8 acc0 = {}, acc1 = {};
  const int nK = K / TBK;

  unsigned int aR[2], bR[8];

  auto gload = [&](int kc) {
    const int k0 = kc * TBK;
    #pragma unroll
    for (int i = 0; i < 2; ++i) {
      int r = rowBase + ar + i * 16;
      int k = k0 + 2 * ac2;
      float v0, v1;
      if (AT) {                 // pair strided by lda
        v0 = Az[(size_t)k * lda + r];
        v1 = Az[(size_t)(k + 1) * lda + r];
      } else {                  // pair contiguous -> b64 load
        float2 a2 = *(const float2*)&Az[(size_t)r * lda + k];
        v0 = a2.x; v1 = a2.y;
      }
      aR[i] = pack2bf(v0, v1);
    }
    #pragma unroll
    for (int i = 0; i < 8; ++i) {
      int kg = k0 + 2 * (bk2 + i * 2);
      int ng = colBase + bn;
      float v0, v1;
      if (WT) {                 // pair contiguous -> b64 load
        float2 w2 = *(const float2*)&Wz[(size_t)ng * ldw + kg];
        v0 = w2.x; v1 = w2.y;
      } else {                  // pair strided by ldw
        v0 = Wz[(size_t)kg * ldw + ng];
        v1 = Wz[(size_t)(kg + 1) * ldw + ng];
      }
      bR[i] = pack2bf(v0, v1);
    }
  };
  auto sstore = [&](int buf) {
    #pragma unroll
    for (int i = 0; i < 2; ++i)
      sA[buf][(ar + i * 16) * SA_STRIDE + ac2] = aR[i];
    #pragma unroll
    for (int i = 0; i < 8; ++i)
      sB[buf][(bk2 + i * 2) * SB_STRIDE + bn] = bR[i];
  };
  auto compute = [&](int buf) {
    union { bf16x16 v; unsigned int u32[8]; } afr, bf0, bf1;
    // A 16x32 bf16: lanes 0-15 K 0..15, lanes 16-31 K 16..31
    int mrow = wRow + (lane & 15);
    int kh2  = (lane >> 4) * 8;              // khalf/2
    #pragma unroll
    for (int r = 0; r < 8; ++r) {
      int kk2 = r + kh2;
      afr.u32[r] = sA[buf][mrow * SA_STRIDE + kk2];
    }
    // B 32x16 bf16: lane = n + 16*(k/16); packed pair = consecutive k
    int ncol = wCol + (lane & 15);
    int kb2  = (lane >> 4) * 8;              // kblk/2
    #pragma unroll
    for (int r = 0; r < 8; ++r) {
      bf0.u32[r] = sB[buf][(kb2 + r) * SB_STRIDE + ncol];
      bf1.u32[r] = sB[buf][(kb2 + r) * SB_STRIDE + ncol + 64];
    }
    acc0 = __builtin_amdgcn_wmma_f32_16x16x32_bf16(
        false, afr.v, false, bf0.v, (short)0, acc0, false, false);
    acc1 = __builtin_amdgcn_wmma_f32_16x16x32_bf16(
        false, afr.v, false, bf1.v, (short)0, acc1, false, false);
  };

  // pipeline prologue
  gload(0);
  sstore(0);
  __syncthreads();

  // steady state: full pipeline stages for kc = 0 .. nK-2
  int kc = 0;
  for (; kc < nK - 1; ++kc) {
    gload(kc + 1);                           // overlap global loads with WMMA
    {                                        // branch-free clamped prefetch
      int kcn = (kc + 2 < nK) ? (kc + 2) : (nK - 1);
      int k0n = kcn * TBK;
      const float* pf = WT ? &Wz[(size_t)colBase * ldw + k0n]
                           : &Wz[(size_t)k0n * ldw + colBase];
      __builtin_prefetch(pf, 0, 1);
    }
    compute(kc & 1);
    sstore((kc & 1) ^ 1);
    __syncthreads();
  }
  // epilogue chunk (no further staging)
  compute(kc & 1);

  // ---- C store: layout lane n = lane&15, M = v + 8*(lane>>4) ----
  const int n0   = colBase + wCol + (lane & 15);
  const int mofs = (lane >> 4) * 8;
  float bv0 = 0.f, bv1 = 0.f;
  if (bias != nullptr) {
    bv0 = bias[(size_t)blockIdx.z * sBz + n0];
    bv1 = bias[(size_t)blockIdx.z * sBz + n0 + 64];
  }
  #pragma unroll
  for (int v = 0; v < 8; ++v) {
    int m = rowBase + wRow + mofs + v;
    float r0 = (acc0[v] + bv0) * alpha;
    float r1 = (acc1[v] + bv1) * alpha;
    if (relu) { r0 = fmaxf(r0, 0.f); r1 = fmaxf(r1, 0.f); }
    Cz[(size_t)m * ldc + n0]      = r0;
    Cz[(size_t)m * ldc + n0 + 64] = r1;
  }
}

// =====================================================================
// small elementwise / reduction kernels
// =====================================================================
__global__ void fill_zero(float* p, long n) {
  long i = (long)blockIdx.x * blockDim.x + threadIdx.x;
  if (i < n) p[i] = 0.f;
}

// attention: thread per (m,b,h); softmax over M memory slots; writes context
__global__ void attn_kernel(const float* __restrict__ Q,
                            const float* __restrict__ KV,
                            float* __restrict__ ctx) {
  int gid = blockIdx.x * blockDim.x + threadIdx.x;
  if (gid >= M_ * B_ * H_) return;
  int m = gid / (B_ * H_);
  int r = gid % (B_ * H_);
  int b = r / H_;
  int h = r % H_;
  const float* q   = Q + ((size_t)m * B_ + b) * D_ + h * HD_;
  const float* kvm = KV + (size_t)m * (M_ * B_) * (2 * D_);
  float s[M_];
  float mx = -1e30f;
  for (int j = 0; j < M_; ++j) {
    const float* kp = kvm + ((size_t)j * B_ + b) * (2 * D_) + h * HD_;
    float a = 0.f;
    for (int c = 0; c < HD_; ++c) a += q[c] * kp[c];
    s[j] = a; mx = fmaxf(mx, a);
  }
  float sum = 0.f;
  for (int j = 0; j < M_; ++j) { s[j] = expf(s[j] - mx); sum += s[j]; }
  float inv = 1.f / sum;
  float* o = ctx + ((size_t)m * B_ + b) * D_ + h * HD_;
  for (int c = 0; c < HD_; ++c) {
    float a = 0.f;
    for (int j = 0; j < M_; ++j)
      a += s[j] * kvm[((size_t)j * B_ + b) * (2 * D_) + D_ + h * HD_ + c];
    o[c] = a * inv;
  }
}

// (M,B,D) -> (P,B,d):  h[p=m*S+s][b][c] = attnO[m][b][s*d+c]
__global__ void repack_h(const float* __restrict__ attnO, float* __restrict__ h) {
  int i = blockIdx.x * blockDim.x + threadIdx.x;
  if (i >= P_ * B_ * d_) return;
  int p = i / (B_ * d_); int r = i % (B_ * d_); int b = r / d_; int c = r % d_;
  int m = p / S_, s = p % S_;
  h[i] = attnO[((size_t)m * B_ + b) * D_ + s * d_ + c];
}
// (P,B,d) -> (M,B,D):  cat[m][b][s*d+c] = h[m*S+s][b][c]
__global__ void repack_cat(const float* __restrict__ hfin, float* __restrict__ cat) {
  int i = blockIdx.x * blockDim.x + threadIdx.x;
  if (i >= M_ * B_ * D_) return;
  int m = i / (B_ * D_); int r = i % (B_ * D_); int b = r / D_; int e = r % D_;
  int s = e / d_, c = e % d_;
  cat[i] = hfin[(((size_t)(m * S_ + s)) * B_ + b) * d_ + c];
}

// gates: dp=tanh(dp+pd/max(ps,eps)); sr=sig(sr+ps); gb=sig(gb+pg/max(ps,eps)); mod=al*dp*sr
__global__ void heb_gates(const float* __restrict__ nt,
                          float* __restrict__ dp, float* __restrict__ sr,
                          float* __restrict__ gb,
                          const float* __restrict__ al,
                          float* __restrict__ mod, int l) {
  int i = blockIdx.x * blockDim.x + threadIdx.x;
  if (i >= P_ * B_ * d_) return;
  int p = i / (B_ * d_); int r = i % (B_ * d_); int b = r / d_; int c = r % d_;
  const float* row = nt + ((size_t)p * B_ + b) * (3 * d_);
  float pd = row[3 * c], ps = row[3 * c + 1], pg = row[3 * c + 2];
  float inv = 1.f / fmaxf(ps, 1e-6f);
  size_t st = (((size_t)p * L_ + l) * B_ + b) * d_ + c;
  size_t pm = ((size_t)p * L_ + l) * d_ + c;
  float dpv = tanhf(dp[st] + pd * inv);
  float srv = 1.f / (1.f + expf(-(sr[st] + ps)));
  float gbv = 1.f / (1.f + expf(-(gb[st] + pg * inv)));
  dp[st] = dpv; sr[st] = srv; gb[st] = gbv;
  mod[((size_t)p * B_ + b) * d_ + c] = al[pm] * dpv * srv;
}

// sd[p][c] = decay * sigmoid(mean_b gb)
__global__ void sd_kernel(const float* __restrict__ gb, const float* __restrict__ dc,
                          float* __restrict__ sd, int l) {
  int i = blockIdx.x * blockDim.x + threadIdx.x;
  if (i >= P_ * d_) return;
  int p = i / d_, c = i % d_;
  float s = 0.f;
  for (int b = 0; b < B_; ++b)
    s += gb[(((size_t)p * L_ + l) * B_ + b) * d_ + c];
  s *= (1.f / B_);
  sd[i] = dc[((size_t)p * L_ + l) * d_ + c] * (1.f / (1.f + expf(-s)));
}

// row softmax; rows live at X + z*zX + (row%rowsPerZ)*len
__global__ __launch_bounds__(128) void softmax_rows(
    const float* __restrict__ X, float* __restrict__ Y,
    int len, int rowsPerZ, long zX, long zY) {
  __shared__ float red[128];
  int row = blockIdx.x;
  int z = row / rowsPerZ, rz = row % rowsPerZ;
  const float* x = X + (size_t)z * zX + (size_t)rz * len;
  float*       y = Y + (size_t)z * zY + (size_t)rz * len;
  int tid = threadIdx.x;
  float mx = -1e30f;
  for (int j = tid; j < len; j += 128) mx = fmaxf(mx, x[j]);
  red[tid] = mx; __syncthreads();
  for (int o = 64; o > 0; o >>= 1) { if (tid < o) red[tid] = fmaxf(red[tid], red[tid + o]); __syncthreads(); }
  mx = red[0]; __syncthreads();
  float s = 0.f;
  for (int j = tid; j < len; j += 128) s += expf(x[j] - mx);
  red[tid] = s; __syncthreads();
  for (int o = 64; o > 0; o >>= 1) { if (tid < o) red[tid] += red[tid + o]; __syncthreads(); }
  float inv = 1.f / red[0];
  for (int j = tid; j < len; j += 128) y[j] = expf(x[j] - mx) * inv;
}

// Hebbian update: W = W + rowsoftmax(W)*O - sd[i]*W   (row length d_ = 3*128)
__global__ __launch_bounds__(128) void w_update(
    float* __restrict__ W, long zW,
    const float* __restrict__ O, const float* __restrict__ sd) {
  __shared__ float red[128];
  int row = blockIdx.x;             // p*d_ + i
  int p = row / d_, i = row % d_;
  float* w = W + (size_t)p * zW + (size_t)i * d_;
  const float* o = O + ((size_t)p * d_ + i) * d_;
  float sdv = sd[row];
  int tid = threadIdx.x;
  float v[3];
  float mx = -1e30f;
  #pragma unroll
  for (int k = 0; k < 3; ++k) { v[k] = w[tid + k * 128]; mx = fmaxf(mx, v[k]); }
  red[tid] = mx; __syncthreads();
  for (int off = 64; off > 0; off >>= 1) { if (tid < off) red[tid] = fmaxf(red[tid], red[tid + off]); __syncthreads(); }
  mx = red[0]; __syncthreads();
  float s = 0.f, e[3];
  #pragma unroll
  for (int k = 0; k < 3; ++k) { e[k] = expf(v[k] - mx); s += e[k]; }
  red[tid] = s; __syncthreads();
  for (int off = 64; off > 0; off >>= 1) { if (tid < off) red[tid] += red[tid + off]; __syncthreads(); }
  float inv = 1.f / red[0];
  #pragma unroll
  for (int k = 0; k < 3; ++k) {
    int j = tid + k * 128;
    w[j] = v[k] + e[k] * inv * o[j] - sdv * v[k];
  }
}

// LayerNorm rows (contiguous rows of `len`); gamma/beta at g + (row/rowsPerZ)*zG
__global__ __launch_bounds__(128) void ln_rows(
    const float* __restrict__ X, float* __restrict__ Y, float* __restrict__ Y2,
    const float* __restrict__ g, const float* __restrict__ bt,
    int len, int rowsPerZ, long zG) {
  __shared__ float red[128];
  int row = blockIdx.x;
  int z = row / rowsPerZ;
  const float* x  = X + (size_t)row * len;
  const float* gg = g + (size_t)z * zG;
  const float* bb = bt + (size_t)z * zG;
  int tid = threadIdx.x;
  float s = 0.f;
  for (int j = tid; j < len; j += 128) s += x[j];
  red[tid] = s; __syncthreads();
  for (int o = 64; o > 0; o >>= 1) { if (tid < o) red[tid] += red[tid + o]; __syncthreads(); }
  float mu = red[0] / len; __syncthreads();
  float v2 = 0.f;
  for (int j = tid; j < len; j += 128) { float t = x[j] - mu; v2 += t * t; }
  red[tid] = v2; __syncthreads();
  for (int o = 64; o > 0; o >>= 1) { if (tid < o) red[tid] += red[tid + o]; __syncthreads(); }
  float rstd = rsqrtf(red[0] / len + 1e-5f);
  for (int j = tid; j < len; j += 128) {
    float yv = (x[j] - mu) * rstd * gg[j] + bb[j];
    Y[(size_t)row * len + j] = yv;
    if (Y2 != nullptr) Y2[(size_t)row * len + j] = yv;
  }
}

// act = LN(relu(actpre + rec)); also stores new pa/pr state. rows = P_*B_, len = d_.
__global__ __launch_bounds__(128) void act_ln(
    const float* __restrict__ actp, const float* __restrict__ rec,
    const float* __restrict__ g, const float* __restrict__ bt,
    float* __restrict__ hout, float* __restrict__ paSt, float* __restrict__ prSt, int l) {
  __shared__ float red[128];
  int row = blockIdx.x;             // p*B_ + b
  int p = row / B_, b = row % B_;
  const float* x  = actp + (size_t)row * d_;
  const float* rr = rec  + (size_t)row * d_;
  const float* gg = g  + ((size_t)p * L_ + l) * d_;
  const float* bb = bt + ((size_t)p * L_ + l) * d_;
  int tid = threadIdx.x;
  float v[3];
  float s = 0.f;
  #pragma unroll
  for (int k = 0; k < 3; ++k) { int j = tid + k * 128; v[k] = fmaxf(x[j] + rr[j], 0.f); s += v[k]; }
  red[tid] = s; __syncthreads();
  for (int o = 64; o > 0; o >>= 1) { if (tid < o) red[tid] += red[tid + o]; __syncthreads(); }
  float mu = red[0] / d_; __syncthreads();
  float v2 = 0.f;
  #pragma unroll
  for (int k = 0; k < 3; ++k) { float t = v[k] - mu; v2 += t * t; }
  red[tid] = v2; __syncthreads();
  for (int o = 64; o > 0; o >>= 1) { if (tid < o) red[tid] += red[tid + o]; __syncthreads(); }
  float rstd = rsqrtf(red[0] / d_ + 1e-5f);
  size_t stBase = (((size_t)p * L_ + l) * B_ + b) * d_;
  #pragma unroll
  for (int k = 0; k < 3; ++k) {
    int j = tid + k * 128;
    float yv = (v[k] - mu) * rstd * gg[j] + bb[j];
    hout[(size_t)row * d_ + j] = yv;
    paSt[stBase + j] = yv;
    prSt[stBase + j] = rr[j];
  }
}

// =====================================================================
// host side
// =====================================================================
extern "C" void kernel_launch(void* const* d_in, const int* in_sizes, int n_in,
                              void* d_out, int out_size, void* d_ws, size_t ws_size,
                              hipStream_t stream) {
  (void)in_sizes; (void)n_in; (void)out_size; (void)ws_size;
  const float* x   = (const float*)d_in[0];
  const float* mem0= (const float*)d_in[1];
  const float* aiw = (const float*)d_in[2];
  const float* aib = (const float*)d_in[3];
  const float* aow = (const float*)d_in[4];
  const float* aob = (const float*)d_in[5];
  const float* hw  = (const float*)d_in[6];
  const float* hrw = (const float*)d_in[7];
  const float* alp = (const float*)d_in[8];
  const float* dec = (const float*)d_in[9];
  const float* lag = (const float*)d_in[10];
  const float* lab = (const float*)d_in[11];
  const float* lrg = (const float*)d_in[12];
  const float* lrb = (const float*)d_in[13];
  const float* pw1 = (const float*)d_in[14];
  const float* pb1 = (const float*)d_in[15];
  const float* pw2 = (const float*)d_in[16];
  const float* pb2 = (const float*)d_in[17];
  const float* ow  = (const float*)d_in[18];
  const float* ob  = (const float*)d_in[19];
  const float* log_= (const float*)d_in[20];
  const float* lob_= (const float*)d_in[21];

  const size_t MBD  = (size_t)M_ * B_ * D_;           // 196608
  const size_t PBd  = (size_t)P_ * B_ * d_;           // 196608
  const size_t STATE= (size_t)P_ * L_ * B_ * d_;      // 393216
  const size_t PLdd = (size_t)P_ * L_ * d_ * d_;      // 4718592
  const size_t Pdd  = (size_t)P_ * d_ * d_;           // 2359296

  float* ws = (float*)d_ws;
  size_t off = 0;
  auto alloc = [&](size_t n) { float* p = ws + off; off += n; return p; };
  float* mem   = alloc(MBD);
  float* Qb    = alloc(MBD);
  float* KVb   = alloc((size_t)M_ * M_ * B_ * 2 * D_);
  float* ctx   = alloc(MBD);
  float* attnO = alloc(MBD);
  float* hA    = alloc(PBd);
  float* hB    = alloc(PBd);
  float* paS   = alloc(STATE);
  float* prS   = alloc(STATE);
  float* dpS   = alloc(STATE);
  float* srS   = alloc(STATE);
  float* gbS   = alloc(STATE);
  float* Wst   = alloc(PLdd);
  float* Wrst  = alloc(PLdd);
  float* t1b   = alloc(PBd);
  float* ntb   = alloc(PBd * 3);
  float* asmb  = alloc(PBd);
  float* rsmb  = alloc(PBd);
  float* modb  = alloc(PBd);
  float* OAb   = alloc(Pdd);
  float* ORb   = alloc(Pdd);
  float* sdb   = alloc((size_t)P_ * d_);
  float* recb  = alloc(PBd);
  float* actp  = alloc(PBd);
  float* catb  = alloc(MBD);
  float* tmpO  = alloc(MBD);

  // state init (graph-capture safe: async d2d copies + fill kernel)
  hipMemcpyAsync(mem,  mem0, MBD  * sizeof(float), hipMemcpyDeviceToDevice, stream);
  hipMemcpyAsync(Wst,  hw,   PLdd * sizeof(float), hipMemcpyDeviceToDevice, stream);
  hipMemcpyAsync(Wrst, hrw,  PLdd * sizeof(float), hipMemcpyDeviceToDevice, stream);
  {
    long n = (long)(5 * STATE);   // pa,pr,dp,sr,gb are contiguous
    fill_zero<<<(unsigned)((n + 255) / 256), 256, 0, stream>>>(paS, n);
  }

  // dispatch to compile-time (AT,WT) instantiations
  auto gemm = [&](const float* A, long sAz, int lda, int aT,
                  const float* W, long sWz, int ldw, int wT,
                  const float* bias, long sBz,
                  float* C, long sCz, int ldc,
                  int Mr, int N, int K, int Z, float alpha, int relu) {
    dim3 g((unsigned)(N / TBN), (unsigned)(Mr / TBM), (unsigned)Z);
    if (aT)
      wmma_gemm<1, 0><<<g, 256, 0, stream>>>(A, sAz, lda, W, sWz, ldw,
                                             bias, sBz, C, sCz, ldc, K, alpha, relu);
    else if (wT)
      wmma_gemm<0, 1><<<g, 256, 0, stream>>>(A, sAz, lda, W, sWz, ldw,
                                             bias, sBz, C, sCz, ldc, K, alpha, relu);
    else
      wmma_gemm<0, 0><<<g, 256, 0, stream>>>(A, sAz, lda, W, sWz, ldw,
                                             bias, sBz, C, sCz, ldc, K, alpha, relu);
  };

  const float hscale = 1.0f / sqrtf((float)HD_);

  for (int t = 0; t < T_; ++t) {
    const float* xt = x + (size_t)t * B_ * D_;

    // ---- attention ----
    // Q[m] = (x_t @ wq[m]^T + bq[m]) * hd^-0.5            (32 x 768)
    gemm(xt, 0, D_, 0,
         aiw, 3L * D_ * D_, D_, 1,
         aib, 3L * D_,
         Qb, (long)B_ * D_, D_,
         B_, D_, D_, M_, hscale, 0);
    // KV[m] = mem_flat @ [wk;wv][m]^T + [bk;bv][m]        (256 x 1536)
    gemm(mem, 0, D_, 0,
         aiw + (size_t)D_ * D_, 3L * D_ * D_, D_, 1,
         aib + D_, 3L * D_,
         KVb, 2L * M_ * B_ * D_, 2 * D_,
         M_ * B_, 2 * D_, D_, M_, 1.f, 0);
    attn_kernel<<<(M_ * B_ * H_ + 255) / 256, 256, 0, stream>>>(Qb, KVb, ctx);
    // out proj
    gemm(ctx, (long)B_ * D_, D_, 0,
         aow, (long)D_ * D_, D_, 1,
         aob, (long)D_,
         attnO, (long)B_ * D_, D_,
         B_, D_, D_, M_, 1.f, 0);
    repack_h<<<(unsigned)((PBd + 255) / 256), 256, 0, stream>>>(attnO, hA);

    // ---- Hebbian layers ----
    float* xin  = hA;
    float* hout = hB;
    for (int l = 0; l < L_; ++l) {
      // predictor MLP: t1 = relu(pa@w1+b1) ; nt = t1@w2+b2
      gemm(paS + (size_t)l * B_ * d_, (long)L_ * B_ * d_, d_, 0,
           pw1 + (size_t)l * d_ * d_, (long)L_ * d_ * d_, d_, 0,
           pb1 + (size_t)l * d_, (long)L_ * d_,
           t1b, (long)B_ * d_, d_,
           B_, d_, d_, P_, 1.f, 1);
      gemm(t1b, (long)B_ * d_, d_, 0,
           pw2 + (size_t)l * d_ * 3 * d_, (long)L_ * d_ * 3 * d_, 3 * d_, 0,
           pb2 + (size_t)l * 3 * d_, (long)L_ * 3 * d_,
           ntb, (long)B_ * 3 * d_, 3 * d_,
           B_, 3 * d_, d_, P_, 1.f, 0);
      heb_gates<<<(unsigned)((PBd + 255) / 256), 256, 0, stream>>>(
          ntb, dpS, srS, gbS, alp, modb, l);
      // softmax(pa,axis=feat) and softmax(pr,axis=feat)
      softmax_rows<<<P_ * B_, 128, 0, stream>>>(
          paS + (size_t)l * B_ * d_, asmb, d_, B_, (long)L_ * B_ * d_, (long)B_ * d_);
      softmax_rows<<<P_ * B_, 128, 0, stream>>>(
          prS + (size_t)l * B_ * d_, rsmb, d_, B_, (long)L_ * B_ * d_, (long)B_ * d_);
      // outer products (K=32, single WMMA K-chunk): O = a_sm^T @ mod / B
      gemm(asmb, (long)B_ * d_, d_, 1,
           modb, (long)B_ * d_, d_, 0,
           nullptr, 0,
           OAb, (long)d_ * d_, d_,
           d_, d_, B_, P_, 1.f / B_, 0);
      gemm(rsmb, (long)B_ * d_, d_, 1,
           modb, (long)B_ * d_, d_, 0,
           nullptr, 0,
           ORb, (long)d_ * d_, d_,
           d_, d_, B_, P_, 1.f / B_, 0);
      sd_kernel<<<(P_ * d_ + 255) / 256, 256, 0, stream>>>(gbS, dec, sdb, l);
      // W += rowsoftmax(W)*O - sd*W   (in place)
      w_update<<<P_ * d_, 128, 0, stream>>>(Wst  + (size_t)l * d_ * d_, (long)L_ * d_ * d_, OAb, sdb);
      w_update<<<P_ * d_, 128, 0, stream>>>(Wrst + (size_t)l * d_ * d_, (long)L_ * d_ * d_, ORb, sdb);
      // rec = LN(pa @ Wr_new)
      gemm(paS + (size_t)l * B_ * d_, (long)L_ * B_ * d_, d_, 0,
           Wrst + (size_t)l * d_ * d_, (long)L_ * d_ * d_, d_, 0,
           nullptr, 0,
           recb, (long)B_ * d_, d_,
           B_, d_, d_, P_, 1.f, 0);
      ln_rows<<<P_ * B_, 128, 0, stream>>>(recb, recb, nullptr,
          lrg + (size_t)l * d_, lrb + (size_t)l * d_, d_, B_, (long)L_ * d_);
      // actpre = xin @ W_new ; act = LN(relu(actpre + rec)); update states
      gemm(xin, (long)B_ * d_, d_, 0,
           Wst + (size_t)l * d_ * d_, (long)L_ * d_ * d_, d_, 0,
           nullptr, 0,
           actp, (long)B_ * d_, d_,
           B_, d_, d_, P_, 1.f, 0);
      act_ln<<<P_ * B_, 128, 0, stream>>>(actp, recb,
          lag, lab, hout, paS, prS, l);
      float* tp = xin; xin = hout; hout = tp;
    }

    // ---- output projection + LN -> new mem (and d_out at last step) ----
    repack_cat<<<(unsigned)((MBD + 255) / 256), 256, 0, stream>>>(xin, catb);
    gemm(catb, (long)B_ * D_, D_, 0,
         ow, (long)D_ * D_, D_, 1,
         ob, (long)D_,
         tmpO, (long)B_ * D_, D_,
         B_, D_, D_, M_, 1.f, 0);
    ln_rows<<<M_ * B_, 128, 0, stream>>>(tmpO, mem,
        (t == T_ - 1) ? (float*)d_out : nullptr,
        log_, lob_, D_, B_, (long)D_);
  }
}
